// DispAgg_25477746000183
// MI455X (gfx1250) — compile-verified, matrix operand order checked by reading, not run on hardware
//
#include <hip/hip_runtime.h>
#include <hip/hip_bf16.h>
#include <math.h>

typedef _Float16 f16;
typedef _Float16 v16h __attribute__((ext_vector_type(16)));
typedef float    v8f  __attribute__((ext_vector_type(8)));

#define D0   49          // low-res disparity
#define HS   32          // low-res H
#define WS   64          // low-res W
#define DP   193         // MAXDISP+1
#define HH   256
#define WW   512
#define NPIX (HH*WW)     // 131072
#define NLOW (D0*HS*WS)  // 100352
#define NVOL (DP*NPIX)   // 25296896

// ---------------------------------------------------------------------------
// K1: 3x3x3 conv, 32ch -> 1ch, via v_wmma_f32_16x16x32_f16.
// GEMM view: out[16 pts] = W[1 x 864] * im2col[864 x 16pts], K chunked per tap
// (32 channels per WMMA). A-matrix rows all replicated with the tap's weights;
// result read from row 0 (acc VGPR0, lanes 0..15).
// ---------------------------------------------------------------------------
__global__ void k_conv_wmma(const float* __restrict__ x,
                            const float* __restrict__ w,
                            float* __restrict__ cout) {
  __shared__ f16 wl[32 * 27];
  for (int t = threadIdx.x; t < 32 * 27; t += blockDim.x)
    wl[t] = (f16)w[t];                 // conv_w is [32][27] flat (OIDHW, O=1)
  __syncthreads();

  const int wave = (int)((blockIdx.x * blockDim.x + threadIdx.x) >> 5);
  const int lane = threadIdx.x & 31;   // wave32

  // 6272 runs of 16 points along W: run -> (d, h, w0)
  const int w0 = (wave & 3) * 16;
  const int h  = (wave >> 2) & 31;
  const int d  = wave >> 7;

  const int half = (lane >> 4) & 1;
  const int n    = lane & 15;          // B column / output point within run
  const int cA   = half * 8;           // channels for v16h elems 0..7
  const int cB   = 16 + half * 8;      // channels for v16h elems 8..15

  v8f acc = {};
  for (int t = 0; t < 27; ++t) {
    const int kd = t / 9 - 1;
    const int kh = (t / 3) % 3 - 1;
    const int kw = t % 3 - 1;

    v16h a, b;
#pragma unroll
    for (int i = 0; i < 8; ++i) {      // all 16 A rows identical
      a[i]     = wl[(cA + i) * 27 + t];
      a[8 + i] = wl[(cB + i) * 27 + t];
    }
    const int dd = d + kd, hh = h + kh, xw = w0 + n + kw;
    const bool ok = (dd >= 0) && (dd < D0) && (hh >= 0) && (hh < HS) &&
                    (xw >= 0) && (xw < WS);
    const int base = dd * (HS * WS) + hh * WS + xw;
#pragma unroll
    for (int i = 0; i < 8; ++i) {
      b[i]     = ok ? (f16)x[(cA + i) * NLOW + base] : (f16)0.f;
      b[8 + i] = ok ? (f16)x[(cB + i) * NLOW + base] : (f16)0.f;
    }
    acc = __builtin_amdgcn_wmma_f32_16x16x32_f16(
        /*neg_a=*/false, a, /*neg_b=*/false, b,
        /*c_mod=*/(short)0, acc, /*reuse_a=*/false, /*reuse_b=*/false);
  }
  if (lane < 16)                        // row M=0 lives in acc[0], lanes 0..15
    cout[d * (HS * WS) + h * WS + w0 + lane] = acc[0];
}

// ---------------------------------------------------------------------------
// K2: per-pixel inverse L1 norm of guidance over 75 channels
// ---------------------------------------------------------------------------
__global__ void k_invsum(const float* __restrict__ lg1, float* __restrict__ inv) {
  const int p = blockIdx.x * blockDim.x + threadIdx.x;
  if (p >= NPIX) return;
  float s = 0.f;
  for (int ch = 0; ch < 75; ++ch) s += fabsf(lg1[ch * NPIX + p]);
  inv[p] = 1.f / fmaxf(s, 1e-12f);
}

// ---------------------------------------------------------------------------
// K3: trilinear upsample [49,32,64] -> [193,256,512], half-pixel centers,
// edge clamp; store f16 so both full-res volumes stay L2-resident (192MB L2).
// ---------------------------------------------------------------------------
__global__ void k_upsample(const float* __restrict__ c, f16* __restrict__ up) {
  const int idx = blockIdx.x * blockDim.x + threadIdx.x;   // NVOL = 98816*256
  const int xx = idx & (WW - 1);
  const int yy = (idx >> 9) & (HH - 1);
  const int dd = idx >> 17;

  float fz = (dd + 0.5f) * ((float)D0 / (float)DP) - 0.5f;
  float fy = (yy + 0.5f) * 0.125f - 0.5f;
  float fx = (xx + 0.5f) * 0.125f - 0.5f;
  fz = fminf(fmaxf(fz, 0.f), (float)(D0 - 1));
  fy = fminf(fmaxf(fy, 0.f), (float)(HS - 1));
  fx = fminf(fmaxf(fx, 0.f), (float)(WS - 1));
  const int z0 = (int)fz, y0 = (int)fy, x0 = (int)fx;
  const int z1 = min(z0 + 1, D0 - 1), y1 = min(y0 + 1, HS - 1),
            x1 = min(x0 + 1, WS - 1);
  const float wz = fz - z0, wy = fy - y0, wx = fx - x0;

  auto at = [&](int z, int y, int x) { return c[z * (HS * WS) + y * WS + x]; };
  const float v00 = at(z0, y0, x0) * (1.f - wx) + at(z0, y0, x1) * wx;
  const float v01 = at(z0, y1, x0) * (1.f - wx) + at(z0, y1, x1) * wx;
  const float v10 = at(z1, y0, x0) * (1.f - wx) + at(z1, y0, x1) * wx;
  const float v11 = at(z1, y1, x0) * (1.f - wx) + at(z1, y1, x1) * wx;
  const float v0 = v00 * (1.f - wy) + v01 * wy;
  const float v1 = v10 * (1.f - wy) + v11 * wy;
  up[idx] = (f16)(v0 * (1.f - wz) + v1 * wz);
}

// ---------------------------------------------------------------------------
// K4: one LGA pass. Block = (64 x, 4 d), grid = (8, 256 y, 49 dz).
// Guidance for the block's 64 pixels cached in LDS (75*64*4B = 19.2 KB) and
// reused by the block's 4 disparity slices of those pixels.
// ---------------------------------------------------------------------------
__global__ void k_lga(const f16* __restrict__ vin, f16* __restrict__ vout,
                      const float* __restrict__ lg1,
                      const float* __restrict__ inv) {
  __shared__ float gl[75 * 64];
  const int xb = blockIdx.x * 64;
  const int y  = blockIdx.y;
  for (int t = threadIdx.y * 64 + threadIdx.x; t < 75 * 64; t += 256) {
    const int ch = t >> 6, i = t & 63;
    gl[t] = lg1[ch * NPIX + y * WW + xb + i] * inv[y * WW + xb + i];
  }
  __syncthreads();

  const int xo = xb + threadIdx.x;
  const int d  = blockIdx.z * 4 + threadIdx.y;
  if (d >= DP) return;

  float acc = 0.f;
#pragma unroll
  for (int i = 0; i < 5; ++i) {
    const int yy = y + i - 2;
    if (yy < 0 || yy >= HH) continue;
#pragma unroll
    for (int j = 0; j < 5; ++j) {
      const int xx = xo + j - 2;
      if (xx < 0 || xx >= WW) continue;
      const int base = yy * WW + xx;
      const float v0 = (float)vin[d * NPIX + base];
      const float vm = (d > 0)      ? (float)vin[(d - 1) * NPIX + base] : 0.f;
      const float vp = (d < DP - 1) ? (float)vin[(d + 1) * NPIX + base] : 0.f;
      const int idx = i * 5 + j;
      acc += gl[idx * 64 + threadIdx.x] * v0 +
             gl[(25 + idx) * 64 + threadIdx.x] * vm +
             gl[(50 + idx) * 64 + threadIdx.x] * vp;
    }
  }
  vout[d * NPIX + y * WW + xo] = (f16)acc;
}

// ---------------------------------------------------------------------------
// K5: fused online softmin over D + disparity regression (t/s). The post-
// softmax L1 normalization in the reference is identity (softmax sums to 1).
// ---------------------------------------------------------------------------
__global__ void k_softargmin(const f16* __restrict__ vol,
                             float* __restrict__ out) {
  const int p = blockIdx.x * blockDim.x + threadIdx.x;
  if (p >= NPIX) return;
  float m = -1e30f, s = 0.f, t = 0.f;
  for (int d = 0; d < DP; ++d) {
    const float v = -(float)vol[d * NPIX + p];
    if (v > m) {
      const float r = __expf(m - v);
      s *= r; t *= r; m = v;
    }
    const float w = __expf(v - m);
    s += w;
    t += w * (float)d;
  }
  out[p] = t / s;
}

// ---------------------------------------------------------------------------
extern "C" void kernel_launch(void* const* d_in, const int* in_sizes, int n_in,
                              void* d_out, int out_size, void* d_ws,
                              size_t ws_size, hipStream_t stream) {
  const float* x      = (const float*)d_in[0];  // [1,32,49,32,64]
  const float* lg1    = (const float*)d_in[1];  // [1,75,256,512]
  const float* conv_w = (const float*)d_in[2];  // [1,32,3,3,3]
  float* out = (float*)d_out;                   // [1,256,512]

  char* ws = (char*)d_ws;
  float* cvol = (float*)ws;                                  // 401,408 B
  float* inv  = (float*)(ws + 401408);                       // 524,288 B
  f16*   up   = (f16*)(ws + 925696);                         // 50,593,792 B
  f16*   lga1 = (f16*)(ws + 925696 + (size_t)NVOL * 2);      // 50,593,792 B

  // conv (WMMA): 6272 waves of 16 output points, 8 waves/block
  k_conv_wmma<<<784, 256, 0, stream>>>(x, conv_w, cvol);
  // guidance L1-norm inverse
  k_invsum<<<(NPIX + 255) / 256, 256, 0, stream>>>(lg1, inv);
  // trilinear upsample to f16 full-res volume (exactly NVOL = 98816*256)
  k_upsample<<<NVOL / 256, 256, 0, stream>>>(cvol, up);
  // two LGA passes, ping-pong between the L2-resident f16 volumes
  dim3 gb(WW / 64, HH, (DP + 3) / 4), bb(64, 4, 1);
  k_lga<<<gb, bb, 0, stream>>>(up, lga1, lg1, inv);
  k_lga<<<gb, bb, 0, stream>>>(lga1, up, lg1, inv);
  // fused softmin + disparity regression
  k_softargmin<<<(NPIX + 255) / 256, 256, 0, stream>>>(up, out);
}
